// MultiheadAttention_55387898249625
// MI455X (gfx1250) — matrix operand order repaired and checked by reading, not compiled, not run on hardware
//
#include <hip/hip_runtime.h>
#include <hip/hip_bf16.h>
#include <stdint.h>

// ---------------------------------------------------------------------------
// MI455X (gfx1250) fused multi-head attention, bf16 WMMA + f32 accumulate.
//   B=4, S=2048, D=512, H=8, Dh=64
// ws (bf16): Qh[H][BS][64] | Kh[H][BS][64] | Vt[H][B][64][S] | heads[BS][512]
//
// Fragment facts (ISA 7.12.2, bf16):
//  A 16x32 : lane l, elems 0..7  = row (l&15), K=(l>=16?8:0)+0..7   (16B run)
//                     elems 8..15 = same row,  K=16+(l>=16?8:0)+0..7 (16B run)
//  B 32x16 : lane l = col (l&15); K order as A; this code uses K=(l>=16?16:0)+e for A and B
//  C/D     : elem r -> row (l>=16?8:0)+r, col l&15
// GEMM kernels: 32x32 register tile per wave (2x2 WMMA) -> 2 LDS loads/WMMA.
// ---------------------------------------------------------------------------

typedef __bf16 bf16_t;
typedef __bf16 v16bf __attribute__((ext_vector_type(16)));
typedef float  v8f   __attribute__((ext_vector_type(8)));

union U16 { uint4 u[2]; v16bf v; };

#define NH   8
#define NB   4
#define SQ   2048
#define DM   512
#define DH   64
#define BS   (NB * SQ)          // 8192

// ---- CDNA5 async global->LDS DMA (ASYNCcnt path, ISA 15.18.3 op 98) -------
__device__ __forceinline__ void async_copy_b128(uint32_t lds_addr, const void* g) {
    asm volatile("global_load_async_to_lds_b128 %0, %1, off"
                 :: "v"(lds_addr), "v"((unsigned long long)(uintptr_t)g)
                 : "memory");
}
__device__ __forceinline__ void wait_async0() {
    asm volatile("s_wait_asynccnt 0x0" ::: "memory");
}
__device__ __forceinline__ uint32_t lds_off(const void* p) {
    // generic LDS pointers carry the wave-relative LDS byte address in [31:0]
    return (uint32_t)(uintptr_t)p;
}

// ===========================================================================
// Kernel 1: fused Q/K/V projection.  X[8192x512] * Wcat[512x1536] -> bf16.
// Block 256 thr (8 waves as 4Mx2N), workgroup tile 128(M) x 64(N), K step 32.
// A row-major in LDS (vector reads); B in fragment order.
// V written TRANSPOSED: Vt[h][b][dh][s].
// ===========================================================================
__global__ __launch_bounds__(256) void mha_qkv_proj(
    const float* __restrict__ q,       // [BS, DM]
    const float* __restrict__ Wq,      // [NH, DM, DH]
    const float* __restrict__ Wk,
    const float* __restrict__ Wv,
    bf16_t* __restrict__ Qh,           // [NH, BS, DH] (pre-scaled by 0.125)
    bf16_t* __restrict__ Kh,           // [NH, BS, DH]
    bf16_t* __restrict__ Vt)           // [NH, NB, DH, SQ]
{
    __shared__ uint4 AsV[512];         // 128 x 32 bf16, row-major (64B rows)
    __shared__ uint4 BfV[256];         // 4 nblk x 32 lane x 16 e (frag order)
    bf16_t* Bf = (bf16_t*)BfV;

    const int tid  = threadIdx.x;
    const int lane = tid & 31;
    const int wave = tid >> 5;
    const int mi   = wave & 3;         // 32-row block
    const int ni   = wave >> 2;        // 32-col block

    const int m0  = blockIdx.x * 128;
    const int n0g = blockIdx.y * 64;

    v8f acc[2][2] = {{{}, {}}, {{}, {}}};

    for (int kt = 0; kt < DM; kt += 32) {
        __syncthreads();
        // ---- stage A: 128x32 f32 -> bf16, 2x ds_store_b128 per thread ----
        #pragma unroll
        for (int t2 = 0; t2 < 2; ++t2) {
            int idx = tid + t2 * 256;
            int row = idx >> 2;
            int kc  = (idx & 3) * 8;
            const float4* src = (const float4*)(q + (size_t)(m0 + row) * DM + kt + kc);
            float4 f0 = src[0], f1 = src[1];
            float tmp[8] = {f0.x, f0.y, f0.z, f0.w, f1.x, f1.y, f1.z, f1.w};
            union { bf16_t h[8]; uint4 u; } pk;
            #pragma unroll
            for (int i = 0; i < 8; ++i) pk.h[i] = (bf16_t)tmp[i];
            AsV[idx] = pk.u;
        }
        // ---- stage B: 32x64, two float4 loads, 8 frag-order writes ----
        {
            int krow = tid >> 3;           // K 0..31
            int c8   = (tid & 7) * 8;      // col 0..56
            int cg0  = n0g + c8;
            int mat  = cg0 >> 9;           // 0=Q,1=K,2=V
            int head = (cg0 & 511) >> 6;
            int cc   = cg0 & 63;
            const float* W = (mat == 0) ? Wq : ((mat == 1) ? Wk : Wv);
            const float4* wp = (const float4*)(W + (size_t)head * DM * DH
                                                 + (size_t)(kt + krow) * DH + cc);
            float4 w0 = wp[0], w1 = wp[1];
            float tmp[8] = {w0.x, w0.y, w0.z, w0.w, w1.x, w1.y, w1.z, w1.w};
            int lhi = (krow & 16) ? 16 : 0;
            #pragma unroll
            for (int i = 0; i < 8; ++i) {
                int col = c8 + i;
                Bf[(col >> 4) * 512 + ((col & 15) + lhi) * 16 + (krow & 15)] =
                    (bf16_t)tmp[i];
            }
        }
        __syncthreads();

        // ---- fragments: 2 A + 2 B (2x b128 each), 4 WMMAs ----
        U16 a[2], b[2];
        #pragma unroll
        for (int s = 0; s < 2; ++s) {
            int a4 = (mi * 32 + s * 16 + (lane & 15)) * 4 + ((lane >= 16) ? 2 : 0);
            a[s].u[0] = AsV[a4];
            a[s].u[1] = AsV[a4 + 1];
        }
        #pragma unroll
        for (int t = 0; t < 2; ++t) {
            int b4 = (ni * 2 + t) * 64 + lane * 2;
            b[t].u[0] = BfV[b4];
            b[t].u[1] = BfV[b4 + 1];
        }
        #pragma unroll
        for (int s = 0; s < 2; ++s)
            #pragma unroll
            for (int t = 0; t < 2; ++t)
                acc[s][t] = __builtin_amdgcn_wmma_f32_16x16x32_bf16(
                    false, a[s].v, false, b[t].v, (short)0, acc[s][t], false, false);
    }

    // ---- epilogue: scatter bf16 (V transposed) ----
    #pragma unroll
    for (int s = 0; s < 2; ++s)
        #pragma unroll
        for (int t = 0; t < 2; ++t) {
            int cg0  = n0g + (ni * 2 + t) * 16;
            int mat  = cg0 >> 9;
            int head = (cg0 & 511) >> 6;
            int cc   = (cg0 & 63) + (lane & 15);
            #pragma unroll
            for (int r = 0; r < 8; ++r) {
                int rowg = m0 + mi * 32 + s * 16 + ((lane >= 16) ? 8 : 0) + r;
                float v  = acc[s][t][r];
                if (mat == 0) {
                    Qh[((size_t)head * BS + rowg) * DH + cc] = (bf16_t)(v * 0.125f);
                } else if (mat == 1) {
                    Kh[((size_t)head * BS + rowg) * DH + cc] = (bf16_t)v;
                } else {
                    int bidx = rowg >> 11, srow = rowg & 2047;
                    Vt[(((size_t)head * NB + bidx) * DH + cc) * SQ + srow] = (bf16_t)v;
                }
            }
        }
}

// ===========================================================================
// Kernel 2: fused flash attention.  Block = 8 waves x 16 q-rows, barrier-free:
// K and V fragments are contiguous global b128 loads (V pre-transposed);
// only P takes an LDS round-trip (per-wave buffer, closed-form swizzle).
// ===========================================================================
__global__ __launch_bounds__(256) void mha_flash_attn(
    const bf16_t* __restrict__ Qh,     // [NH, BS, DH] pre-scaled
    const bf16_t* __restrict__ Kh,     // [NH, BS, DH]
    const bf16_t* __restrict__ Vt,     // [NH, NB, DH, SQ]
    const uint8_t* __restrict__ mask,  // [NB, SQ, SQ] bool bytes
    bf16_t* __restrict__ heads)        // [BS, NH*DH]
{
    __shared__ uint4 PlV[8 * 64];      // per-wave 16x32 bf16 P, fragment order
    bf16_t* Pl = (bf16_t*)PlV;

    const int tid  = threadIdx.x;
    const int lane = tid & 31;
    const int wave = tid >> 5;

    const int blk  = blockIdx.x;       // 512 blocks
    const int qblk = blk & 15;
    const int b    = (blk >> 4) & 3;
    const int h    = blk >> 6;

    const int qbase = qblk * 128 + wave * 16;
    const int n     = lane & 15;
    const int lhi   = (lane >= 16) ? 1 : 0;

    const uint4* Qv = (const uint4*)(Qh + ((size_t)h * BS + (size_t)b * SQ) * DH);
    const uint4* Kv = (const uint4*)(Kh + ((size_t)h * BS + (size_t)b * SQ) * DH);
    const uint4* Vv = (const uint4*)(Vt + (((size_t)h * NB + b) * DH) * SQ);
    const uint8_t* mp = mask + (size_t)b * SQ * SQ;

    // ---- Q fragments: 4x global b128 ----
    v16bf aq0, aq1;
    {
        size_t q4 = (size_t)(qbase + n) * 8 + 2 * lhi;
        U16 t0, t1;
        t0.u[0] = Qv[q4];     t0.u[1] = Qv[q4 + 1];
        t1.u[0] = Qv[q4 + 4]; t1.u[1] = Qv[q4 + 5];
        aq0 = t0.v; aq1 = t1.v;
    }

    v8f o[4] = {{}, {}, {}, {}};
    float mrow[8], lrow[8];
    #pragma unroll
    for (int r = 0; r < 8; ++r) { mrow[r] = -3.0e38f; lrow[r] = 0.0f; }

    for (int kb = 0; kb < SQ / 32; ++kb) {
        if (kb + 1 < SQ / 32) {        // -> global_prefetch_b8
            __builtin_prefetch(&Kv[(size_t)(kb + 1) * 256 + tid], 0, 0);
            __builtin_prefetch(&Vv[(size_t)(kb + 1) * 4 + (tid & 3) + (size_t)(tid >> 2) * 256], 0, 0);
        }

        // ---- scores S[16x32] = Q * K^T (scale folded into Q), then mask ----
        float sc[2][8];
        #pragma unroll
        for (int kh = 0; kh < 2; ++kh) {
            int krow = kb * 32 + kh * 16 + n;
            U16 t0, t1;
            size_t k4 = (size_t)krow * 8 + 2 * lhi;
            t0.u[0] = Kv[k4];     t0.u[1] = Kv[k4 + 1];      // dh 0..31 slice
            t1.u[0] = Kv[k4 + 4]; t1.u[1] = Kv[k4 + 5];      // dh 32..63 slice
            v8f s = {};
            s = __builtin_amdgcn_wmma_f32_16x16x32_bf16(false, aq0, false, t0.v,
                                                        (short)0, s, false, false);
            s = __builtin_amdgcn_wmma_f32_16x16x32_bf16(false, aq1, false, t1.v,
                                                        (short)0, s, false, false);
            #pragma unroll
            for (int r = 0; r < 8; ++r) {
                int qg  = qbase + ((lane >= 16) ? 8 : 0) + r;
                int key = kb * 32 + kh * 16 + n;
                sc[kh][r] = mp[(size_t)qg * SQ + key] ? s[r] : -1.0e9f;
            }
        }

        // ---- online softmax ----
        float rmax[8];
        #pragma unroll
        for (int r = 0; r < 8; ++r) rmax[r] = fmaxf(sc[0][r], sc[1][r]);
        #pragma unroll
        for (int off = 8; off >= 1; off >>= 1)
            #pragma unroll
            for (int r = 0; r < 8; ++r)
                rmax[r] = fmaxf(rmax[r], __shfl_xor(rmax[r], off, 32));

        float corr[8], psum[8];
        #pragma unroll
        for (int r = 0; r < 8; ++r) {
            float mn = fmaxf(mrow[r], rmax[r]);
            corr[r]  = __expf(mrow[r] - mn);
            mrow[r]  = mn;
            psum[r]  = 0.0f;
        }
        // P in fragment order: lane' = m + kh*16, e' = n
        #pragma unroll
        for (int kh = 0; kh < 2; ++kh)
            #pragma unroll
            for (int r = 0; r < 8; ++r) {
                float p = __expf(sc[kh][r] - mrow[r]);
                psum[r] += p;
                int m = ((lane >= 16) ? 8 : 0) + r;
                Pl[wave * 512 + (m + kh * 16) * 16 + n] =
                    (bf16_t)p;
            }
        #pragma unroll
        for (int off = 8; off >= 1; off >>= 1)
            #pragma unroll
            for (int r = 0; r < 8; ++r)
                psum[r] += __shfl_xor(psum[r], off, 32);
        #pragma unroll
        for (int r = 0; r < 8; ++r) lrow[r] = lrow[r] * corr[r] + psum[r];

        #pragma unroll
        for (int c = 0; c < 4; ++c)
            #pragma unroll
            for (int r = 0; r < 8; ++r) o[c][r] *= corr[r];

        // ---- O += P[16x32] * V[32x64] : P via 2x ds_load_b128, V global ----
        U16 tp;
        tp.u[0] = PlV[wave * 64 + lane * 2];
        tp.u[1] = PlV[wave * 64 + lane * 2 + 1];
        #pragma unroll
        for (int c = 0; c < 4; ++c) {
            int dh = c * 16 + n;
            U16 tv;
            size_t v4 = (size_t)dh * 256 + (size_t)kb * 4 + 2 * lhi;
            tv.u[0] = Vv[v4]; tv.u[1] = Vv[v4 + 1];
            o[c] = __builtin_amdgcn_wmma_f32_16x16x32_bf16(false, tp.v, false, tv.v,
                                                           (short)0, o[c], false, false);
        }
    }

    // ---- normalize, emit merged heads bf16 ----
    float inv[8];
    #pragma unroll
    for (int r = 0; r < 8; ++r) inv[r] = 1.0f / lrow[r];
    #pragma unroll
    for (int c = 0; c < 4; ++c)
        #pragma unroll
        for (int r = 0; r < 8; ++r) {
            int qg = qbase + ((lane >= 16) ? 8 : 0) + r;
            heads[((size_t)b * SQ + qg) * (NH * DH) + h * DH + c * 16 + n] =
                (bf16_t)(o[c][r] * inv[r]);
        }
}

// ===========================================================================
// Kernel 3: output projection.  heads[8192x512](bf16) * Wout[512x512] -> f32.
// Workgroup tile 128x64, 32x32 per wave.  A tile staged with CDNA5 async
// global->LDS DMA (2x b128 per lane, ASYNCcnt).
// ===========================================================================
__global__ __launch_bounds__(256) void mha_out_proj(
    const bf16_t* __restrict__ Hm,     // [BS, 512]
    const float* __restrict__ Wout,    // flat [512, 512]
    float* __restrict__ out)           // [BS, 512]
{
    __shared__ uint4 AsV[512];         // 128 x 32 bf16 row-major
    __shared__ uint4 BfV[256];         // fragment-order B (4 nblk)
    bf16_t* Bf = (bf16_t*)BfV;

    const int tid  = threadIdx.x;
    const int lane = tid & 31;
    const int wave = tid >> 5;
    const int mi   = wave & 3;
    const int ni   = wave >> 2;

    const int m0 = blockIdx.x * 128;
    const int n0 = blockIdx.y * 64;

    v8f acc[2][2] = {{{}, {}}, {{}, {}}};

    for (int kt = 0; kt < 512; kt += 32) {
        __syncthreads();
        // ---- A tile: async DMA, 2 x 16B per lane, row-major ----
        #pragma unroll
        for (int t2 = 0; t2 < 2; ++t2) {
            int idx = tid + t2 * 256;
            int row = idx >> 2;
            int kc  = (idx & 3) * 8;
            async_copy_b128(lds_off(&AsV[idx]),
                            Hm + (size_t)(m0 + row) * 512 + kt + kc);
        }
        // ---- B tile: 32x64 f32 -> bf16, fragment order ----
        {
            int krow = tid >> 3;
            int c8   = (tid & 7) * 8;
            const float4* wp = (const float4*)(Wout + (size_t)(kt + krow) * 512 + n0 + c8);
            float4 w0 = wp[0], w1 = wp[1];
            float tmp[8] = {w0.x, w0.y, w0.z, w0.w, w1.x, w1.y, w1.z, w1.w};
            int lhi = (krow & 16) ? 16 : 0;
            #pragma unroll
            for (int i = 0; i < 8; ++i) {
                int col = c8 + i;
                Bf[(col >> 4) * 512 + ((col & 15) + lhi) * 16 + (krow & 15)] =
                    (bf16_t)tmp[i];
            }
        }
        wait_async0();
        __syncthreads();

        U16 a[2], b[2];
        #pragma unroll
        for (int s = 0; s < 2; ++s) {
            int a4 = (mi * 32 + s * 16 + (lane & 15)) * 4 + ((lane >= 16) ? 2 : 0);
            a[s].u[0] = AsV[a4];
            a[s].u[1] = AsV[a4 + 1];
        }
        #pragma unroll
        for (int t = 0; t < 2; ++t) {
            int b4 = (ni * 2 + t) * 64 + lane * 2;
            b[t].u[0] = BfV[b4];
            b[t].u[1] = BfV[b4 + 1];
        }
        #pragma unroll
        for (int s = 0; s < 2; ++s)
            #pragma unroll
            for (int t = 0; t < 2; ++t)
                acc[s][t] = __builtin_amdgcn_wmma_f32_16x16x32_bf16(
                    false, a[s].v, false, b[t].v, (short)0, acc[s][t], false, false);
    }

    #pragma unroll
    for (int s = 0; s < 2; ++s)
        #pragma unroll
        for (int t = 0; t < 2; ++t)
            #pragma unroll
            for (int r = 0; r < 8; ++r) {
                int rowg = m0 + mi * 32 + s * 16 + ((lane >= 16) ? 8 : 0) + r;
                int colg = n0 + (ni * 2 + t) * 16 + (lane & 15);
                out[(size_t)rowg * 512 + colg] = acc[s][t][r];
            }
}

// ===========================================================================
extern "C" void kernel_launch(void* const* d_in, const int* in_sizes, int n_in,
                              void* d_out, int out_size, void* d_ws, size_t ws_size,
                              hipStream_t stream) {
    const float*   q    = (const float*)d_in[0];
    const float*   Wq   = (const float*)d_in[1];
    const float*   Wk   = (const float*)d_in[2];
    const float*   Wv   = (const float*)d_in[3];
    const float*   Wout = (const float*)d_in[4];
    const uint8_t* mask = (const uint8_t*)d_in[5];   // jnp bool -> 1 byte/elem
    float*         out  = (float*)d_out;

    const size_t qkv_elems = (size_t)NH * BS * DH;   // 4,194,304 each
    bf16_t* Qh = (bf16_t*)d_ws;
    bf16_t* Kh = Qh + qkv_elems;
    bf16_t* Vt = Kh + qkv_elems;                     // transposed V
    bf16_t* Hm = Vt + qkv_elems;                     // BS*512 bf16

    mha_qkv_proj<<<dim3(BS / 128, 1536 / 64), 256, 0, stream>>>(
        q, Wq, Wk, Wv, Qh, Kh, Vt);

    mha_flash_attn<<<NH * NB * (SQ / 128), 256, 0, stream>>>(
        Qh, Kh, Vt, mask, Hm);

    mha_out_proj<<<dim3(BS / 128, 512 / 64), 256, 0, stream>>>(Hm, Wout, out);
}
